// AttentionLayer_41180146434697
// MI455X (gfx1250) — matrix operand order repaired and checked by reading, not compiled, not run on hardware
//
#include <hip/hip_runtime.h>
#include <hip/hip_bf16.h>
#include <math.h>

// ---------------------------------------------------------------------------
// AttentionLayer for MI455X (gfx1250, wave32, WMMA 16x16x32 f16, f32 accum)
// B=4, S=2048, D=1024, H=16, Dh=64.  ~120 GFLOP vs ~0.2GB traffic -> compute
// bound: all matrix math on v_wmma_f32_16x16x32_f16; loads ping-pong
// double-buffered (manual unroll x2 => no register-rotation movs).
// ---------------------------------------------------------------------------

typedef __attribute__((ext_vector_type(16))) _Float16 v16h;
typedef __attribute__((ext_vector_type(8)))  _Float16 v8h;
typedef __attribute__((ext_vector_type(4)))  _Float16 v4h;
typedef __attribute__((ext_vector_type(8)))  float    v8f;

#define BATCH 4
#define SEQ   2048
#define DMODEL 1024
#define HEADS 16
#define DHEAD 64
#define NEG_INF_F (-10000.0f)

static __device__ __forceinline__ v16h cat8(v8h lo, v8h hi) {
    return __builtin_shufflevector(lo, hi, 0, 1, 2, 3, 4, 5, 6, 7,
                                   8, 9, 10, 11, 12, 13, 14, 15);
}

// -------------------------------------------------------------------
// Kernel 1a: fp32 -> f16 conversion of x (8192 x 1024), 4-wide vectorized
// -------------------------------------------------------------------
__global__ __launch_bounds__(256) void cvt_x_kernel(const float* __restrict__ x,
                                                    _Float16* __restrict__ xh) {
    int i = blockIdx.x * blockDim.x + threadIdx.x;   // 2M threads, 4 elems each
    float4 v = ((const float4*)x)[i];
    v4h o;
    o[0] = (_Float16)v.x; o[1] = (_Float16)v.y;
    o[2] = (_Float16)v.z; o[3] = (_Float16)v.w;
    ((v4h*)xh)[i] = o;
}

// -------------------------------------------------------------------
// Kernel 1b: fp32 W [D,N] -> f16 Wt [N,D] via 32x32 LDS tile transpose
// (coalesced read AND write).  Grid: (1024/32)^2 = 1024 blocks, 256 thr.
// -------------------------------------------------------------------
__global__ __launch_bounds__(256) void cvt_wt_kernel(const float* __restrict__ W,
                                                     _Float16* __restrict__ Wt) {
    __shared__ float tile[32][33];                   // +1 pad: no bank conflicts
    const int tx = threadIdx.x & 31;                 // col within tile
    const int tq = threadIdx.x >> 5;                 // 8 row-groups of 4
    const int d0 = (blockIdx.x & 31) * 32;           // source row block (D)
    const int n0 = (blockIdx.x >> 5) * 32;           // source col block (N)
#pragma unroll
    for (int r = 0; r < 4; ++r) {                    // read W[d0+row][n0+tx]
        const int row = tq * 4 + r;
        tile[row][tx] = W[(size_t)(d0 + row) * DMODEL + n0 + tx];
    }
    __syncthreads();
#pragma unroll
    for (int r = 0; r < 4; ++r) {                    // write Wt[n0+row][d0+tx]
        const int row = tq * 4 + r;
        Wt[(size_t)(n0 + row) * DMODEL + d0 + tx] = (_Float16)tile[tx][row];
    }
}

// -------------------------------------------------------------------
// Kernel 2: QKV projection GEMM  [8192 x 1024] @ [1024 x 1024] + bias.
// One wave computes a 16x64 tile (4 accumulators reuse A); 32 WMMA K-steps,
// ping-pong double-buffered (unroll x2, zero rotation movs).
// which = blockIdx.y : 0=Q, 1=K -> [B,H,S,Dh] f16; 2=V -> transposed
// [B,H,Dh,S] f16 so the PV GEMM's B-operand is a contiguous 32B lane read.
// -------------------------------------------------------------------
__global__ __launch_bounds__(128) void qkv_gemm_kernel(
    const _Float16* __restrict__ xh,
    const _Float16* __restrict__ Wqt, const _Float16* __restrict__ Wkt,
    const _Float16* __restrict__ Wvt,
    const float* __restrict__ bq, const float* __restrict__ bk,
    const float* __restrict__ bv,
    _Float16* __restrict__ Qh, _Float16* __restrict__ Kh,
    _Float16* __restrict__ Vt) {

    const int lane = threadIdx.x & 31;
    const int wave = threadIdx.x >> 5;
    const int lm = lane & 15;      // A: row m / B: col n  (within tile)
    const int hh = lane >> 4;      // half-wave selects K chunk

    const int which = blockIdx.y;
    const _Float16* Wt  = (which == 0) ? Wqt : (which == 1) ? Wkt : Wvt;
    const float*    bia = (which == 0) ? bq  : (which == 1) ? bk  : bv;

    const int t  = blockIdx.x * 4 + wave;  // 8192 tiles: 512 m-tiles x 16 n-tiles
    const int m0 = (t >> 4) << 4;
    const int n0 = (t & 15) << 6;

    const _Float16* arow = xh + (size_t)(m0 + lm) * DMODEL;
    const _Float16* brow[4];
#pragma unroll
    for (int c = 0; c < 4; ++c)
        brow[c] = Wt + (size_t)(n0 + c * 16 + lm) * DMODEL + hh * 16;

    // A operand (16x32 f16): lane row = lm; element s -> K = hh*16+s
    // => one contiguous 32B chunk.
    auto loadA = [&](int kk) -> v16h {
        return *(const v16h*)(arow + kk + hh * 16);
    };

    v8f acc[4] = {};
    v16h a0, a1, b0[4], b1[4];

    a0 = loadA(0);
#pragma unroll
    for (int c = 0; c < 4; ++c) b0[c] = *(const v16h*)(brow[c]);

    for (int kk = 0; kk < DMODEL - 64; kk += 64) {
        a1 = loadA(kk + 32);                              // fill set1
#pragma unroll
        for (int c = 0; c < 4; ++c) b1[c] = *(const v16h*)(brow[c] + kk + 32);
#pragma unroll
        for (int c = 0; c < 4; ++c)                       // compute set0
            acc[c] = __builtin_amdgcn_wmma_f32_16x16x32_f16(
                false, a0, false, b0[c], (short)0, acc[c], false, false);
        a0 = loadA(kk + 64);                              // refill set0
#pragma unroll
        for (int c = 0; c < 4; ++c) b0[c] = *(const v16h*)(brow[c] + kk + 64);
#pragma unroll
        for (int c = 0; c < 4; ++c)                       // compute set1
            acc[c] = __builtin_amdgcn_wmma_f32_16x16x32_f16(
                false, a1, false, b1[c], (short)0, acc[c], false, false);
    }
    // tail: set0 = @DMODEL-64; load and compute final two K-steps
    a1 = loadA(DMODEL - 32);
#pragma unroll
    for (int c = 0; c < 4; ++c) b1[c] = *(const v16h*)(brow[c] + DMODEL - 32);
#pragma unroll
    for (int c = 0; c < 4; ++c)
        acc[c] = __builtin_amdgcn_wmma_f32_16x16x32_f16(
            false, a0, false, b0[c], (short)0, acc[c], false, false);
#pragma unroll
    for (int c = 0; c < 4; ++c)
        acc[c] = __builtin_amdgcn_wmma_f32_16x16x32_f16(
            false, a1, false, b1[c], (short)0, acc[c], false, false);

    // Epilogue: + bias, cvt f16, scatter into attention-friendly layouts.
#pragma unroll
    for (int c = 0; c < 4; ++c) {
        const int n  = n0 + c * 16 + lm;
        const float bb = bia[n];
        const int h  = n >> 6;
        const int dh = n & 63;
#pragma unroll
        for (int r = 0; r < 8; ++r) {
            // C layout: element (m = r + 8*hh, col = lm) per 16x16 tile
            const int grow = m0 + r + 8 * hh;
            const int b = grow >> 11;          // /2048
            const int s = grow & 2047;
            const size_t bh = (size_t)(b * HEADS + h);
            const _Float16 val = (_Float16)(acc[c][r] + bb);
            if (which == 2)
                Vt[bh * DHEAD * SEQ + (size_t)dh * SEQ + s] = val;
            else if (which == 1)
                Kh[(bh * SEQ + s) * DHEAD + dh] = val;
            else
                Qh[(bh * SEQ + s) * DHEAD + dh] = val;
        }
    }
}

// -------------------------------------------------------------------
// Kernel 3: flash attention.  One wave owns 16 queries of one (b,h);
// streams keys in 32-wide blocks: 4 WMMAs S=Q*K^T, online softmax (row
// stats via __shfl_xor within 16-lane halves of the C layout), P relaid
// C->A layout through per-wave LDS fenced with s_wait_dscnt (LDS ops are
// in-order per wave), 4 WMMAs O+=P*V.  K operands ping-pong double-
// buffered across the fence (unroll x2, no rotation movs); V + mask loads
// hoisted so they overlap the score WMMAs and the softmax VALU chain.
// -------------------------------------------------------------------
__global__ __launch_bounds__(128) void flash_attn_kernel(
    const _Float16* __restrict__ Qh, const _Float16* __restrict__ Kh,
    const _Float16* __restrict__ Vt, const int* __restrict__ mask,
    float* __restrict__ out) {

    __shared__ float pls[4][16 * 33];   // per-wave 16x32 P tile, padded stride

    const int lane = threadIdx.x & 31;
    const int wave = threadIdx.x >> 5;
    const int lm = lane & 15;
    const int hh = lane >> 4;

    const int blk = blockIdx.x;          // B*H*(S/64) = 2048
    const int qt  = blk & 31;            // 32 query-tiles of 64
    const int bh  = blk >> 5;            // b*16 + h
    const int b   = bh >> 4;
    const int h   = bh & 15;
    const int q0  = qt * 64 + wave * 16;

    const _Float16* Qb = Qh + (size_t)bh * SEQ * DHEAD;
    const _Float16* Kb = Kh + (size_t)bh * SEQ * DHEAD;
    const _Float16* Vb = Vt + (size_t)bh * DHEAD * SEQ;
    const int* mrow = mask + b * SEQ;
    float* pw = &pls[wave][0];

    // Q as A operand for both K-steps of the Dh=64 reduction.
    v16h qa[2];
#pragma unroll
    for (int ks = 0; ks < 2; ++ks) {
        const _Float16* qp = Qb + (size_t)(q0 + lm) * DHEAD + ks * 32;
        qa[ks] = *(const v16h*)(qp + hh * 16);
    }

    // B operand for scores: lane col = key; 16 contiguous K (=dh) values.
    auto loadK = [&](int k0, v16h kb[2][2]) {
#pragma unroll
        for (int nt = 0; nt < 2; ++nt) {
            const _Float16* kp = Kb + (size_t)(k0 + nt * 16 + lm) * DHEAD;
#pragma unroll
            for (int ks = 0; ks < 2; ++ks)
                kb[nt][ks] = *(const v16h*)(kp + ks * 32 + hh * 16);
        }
    };

    v8f o[4] = {};
    float mst[8], lst[8];
#pragma unroll
    for (int r = 0; r < 8; ++r) { mst[r] = -1.0e30f; lst[r] = 0.0f; }

    // one 32-key step: scores -> online softmax -> LDS relayout -> O update
    auto step = [&](int k0, v16h kb[2][2]) {
        v16h vbuf[4];
#pragma unroll
        for (int tv = 0; tv < 4; ++tv)   // V^T: 32B contiguous along keys
            vbuf[tv] = *(const v16h*)(Vb + (size_t)(tv * 16 + lm) * SEQ
                                      + k0 + hh * 16);
        const float add0 = (1.0f - (float)mrow[k0 + lm]) * NEG_INF_F;
        const float add1 = (1.0f - (float)mrow[k0 + 16 + lm]) * NEG_INF_F;

        v8f sc[2] = {};
#pragma unroll
        for (int nt = 0; nt < 2; ++nt)
#pragma unroll
            for (int ks = 0; ks < 2; ++ks)
                sc[nt] = __builtin_amdgcn_wmma_f32_16x16x32_f16(
                    false, qa[ks], false, kb[nt][ks], (short)0, sc[nt],
                    false, false);

#pragma unroll
        for (int r = 0; r < 8; ++r) {
            const float s0 = sc[0][r] * 0.125f + add0;   // 1/sqrt(64)
            const float s1 = sc[1][r] * 0.125f + add1;
            float mx = fmaxf(s0, s1);
            mx = fmaxf(mx, __shfl_xor(mx, 1));
            mx = fmaxf(mx, __shfl_xor(mx, 2));
            mx = fmaxf(mx, __shfl_xor(mx, 4));
            mx = fmaxf(mx, __shfl_xor(mx, 8));
            const float mnew = fmaxf(mst[r], mx);
            const float alpha = __expf(mst[r] - mnew);
            mst[r] = mnew;
            const float p0 = __expf(s0 - mnew);
            const float p1 = __expf(s1 - mnew);
            float rs = p0 + p1;
            rs += __shfl_xor(rs, 1);
            rs += __shfl_xor(rs, 2);
            rs += __shfl_xor(rs, 4);
            rs += __shfl_xor(rs, 8);
            lst[r] = lst[r] * alpha + rs;
#pragma unroll
            for (int tv = 0; tv < 4; ++tv) o[tv][r] *= alpha;
            sc[0][r] = p0;
            sc[1][r] = p1;
        }

        // relayout P: C-layout (row m = r+8*hh, col lm) -> A-operand layout
#pragma unroll
        for (int r = 0; r < 8; ++r) {
            const int m = r + 8 * hh;
            pw[m * 33 + lm]      = sc[0][r];
            pw[m * 33 + 16 + lm] = sc[1][r];
        }
        __asm__ volatile("s_wait_dscnt 0" ::: "memory");  // in-wave LDS fence
        v16h pa;
#pragma unroll
        for (int e = 0; e < 8; ++e)
            pa[e] = (_Float16)pw[lm * 33 + hh * 16 + e];
#pragma unroll
        for (int e = 0; e < 8; ++e)
            pa[8 + e] = (_Float16)pw[lm * 33 + hh * 16 + 8 + e];
        __asm__ volatile("s_wait_dscnt 0" ::: "memory");  // reads before rewrite

#pragma unroll
        for (int tv = 0; tv < 4; ++tv)
            o[tv] = __builtin_amdgcn_wmma_f32_16x16x32_f16(
                false, pa, false, vbuf[tv], (short)0, o[tv], false, false);
    };

    // ping-pong over 64-key super-iterations (no K-buffer rotation movs)
    v16h kbA[2][2], kbB[2][2];
    loadK(0, kbA);
    int k0 = 0;
    for (; k0 < SEQ - 64; k0 += 64) {
        loadK(k0 + 32, kbB);
        step(k0, kbA);
        loadK(k0 + 64, kbA);
        step(k0 + 32, kbB);
    }
    loadK(k0 + 32, kbB);     // final super-iteration
    step(k0, kbA);
    step(k0 + 32, kbB);

    // ---- epilogue: normalize and write [B,S,H*Dh] fp32 ----
#pragma unroll
    for (int r = 0; r < 8; ++r) {
        const float inv = 1.0f / lst[r];
        const int m = r + 8 * hh;
        const size_t row = (size_t)b * SEQ + q0 + m;
#pragma unroll
        for (int tv = 0; tv < 4; ++tv)
            out[row * (HEADS * DHEAD) + h * DHEAD + tv * 16 + lm] = o[tv][r] * inv;
    }
}

// -------------------------------------------------------------------
// Host launcher
// -------------------------------------------------------------------
extern "C" void kernel_launch(void* const* d_in, const int* in_sizes, int n_in,
                              void* d_out, int out_size, void* d_ws, size_t ws_size,
                              hipStream_t stream) {
    const float* x    = (const float*)d_in[0];
    const int*   mask = (const int*)d_in[1];
    const float* Wq   = (const float*)d_in[2];
    const float* bq   = (const float*)d_in[3];
    const float* Wk   = (const float*)d_in[4];
    const float* bk   = (const float*)d_in[5];
    const float* Wv   = (const float*)d_in[6];
    const float* bv   = (const float*)d_in[7];
    float* out = (float*)d_out;

    // workspace carve-up (f16 elements): ~70MB total
    _Float16* ws  = (_Float16*)d_ws;
    const size_t XN = (size_t)BATCH * SEQ * DMODEL;      // 8M
    const size_t WN = (size_t)DMODEL * DMODEL;           // 1M
    _Float16* xh  = ws;
    _Float16* Wqt = xh  + XN;
    _Float16* Wkt = Wqt + WN;
    _Float16* Wvt = Wkt + WN;
    _Float16* Qh  = Wvt + WN;
    _Float16* Kh  = Qh  + XN;
    _Float16* Vt  = Kh  + XN;

    // 1) conversions
    cvt_x_kernel<<<(int)(XN / 4 / 256), 256, 0, stream>>>(x, xh);
    cvt_wt_kernel<<<1024, 256, 0, stream>>>(Wq, Wqt);
    cvt_wt_kernel<<<1024, 256, 0, stream>>>(Wk, Wkt);
    cvt_wt_kernel<<<1024, 256, 0, stream>>>(Wv, Wvt);

    // 2) QKV projections: 8192 tiles / 4 waves-per-block, y = Q/K/V
    dim3 ggrid(2048, 3, 1);
    qkv_gemm_kernel<<<ggrid, 128, 0, stream>>>(xh, Wqt, Wkt, Wvt, bq, bk, bv,
                                               Qh, Kh, Vt);

    // 3) flash attention: B*H*(S/64) = 2048 blocks x 4 waves
    flash_attn_kernel<<<2048, 128, 0, stream>>>(Qh, Kh, Vt, mask, out);
}
